// LSTM_18476949307655
// MI455X (gfx1250) — matrix operand order repaired and checked by reading, not compiled, not run on hardware
//
#include <hip/hip_runtime.h>
#include <hip/hip_bf16.h>

// ---------------------------------------------------------------------------
// Persistent LSTM on MI455X (gfx1250): B=128, T=1024, D=64, H=256 (4H=1024), C=6
//   Grid = 64 persistent WGs: (batch tile bt 0..7) x (column group g 0..7).
//   WG (bt,g) owns h columns [32g, 32g+32) of batch rows [16bt, 16bt+16).
//   Its Wh slice (64 KB bf16, WMMA B-fragments) lives in LDS for all 1024 steps.
//   Per step: h exchanged through L2 in A-fragment layout + atomic barrier.
//   bf16 WMMA 16x16x32, f32 accumulation.
// ---------------------------------------------------------------------------

typedef __attribute__((ext_vector_type(16))) __bf16 v16bf;
typedef __attribute__((ext_vector_type(8)))  float  v8f;

typedef __attribute__((address_space(1))) __bf16 gbf;
typedef __attribute__((address_space(1))) v16bf  gv16bf;

#define B_   128
#define T_   1024
#define D_   64
#define H_   256
#define G_   1024   // 4*H
#define C_   6

// workspace layout, in __bf16 elements
#define XP_OFF    0u
#define XP_ELEMS  (8u * 1024u * 2u * 512u)       // x A-fragments: 16 MB
#define WXP_OFF   (XP_OFF + XP_ELEMS)
#define WXP_ELEMS (2u * 64u * 512u)              // Wx B-fragments [kt][nt]
#define WHP_OFF   (WXP_OFF + WXP_ELEMS)
#define WHP_ELEMS (8u * 8u * 8u * 512u)          // Wh B-fragments [g][kt][q*2+j]
#define WOP_OFF   (WHP_OFF + WHP_ELEMS)
#define WOP_ELEMS (8u * 512u)                    // Wout B-fragments (N padded to 16)
#define HX_OFF    (WOP_OFF + WOP_ELEMS)
#define HX_ELEMS  (8u * 2u * 8u * 512u)          // h exchange: [bt][parity][kt] A-frags
#define CNT_BYTE  ((HX_OFF + HX_ELEMS) * 2u)     // 8 x u32 barrier counters

// Fragment element decoders (ISA 7.12.2):
//   B 32x16 bf16:  lane L, elem i:  k = (i/8)*16 + (L/16)*8 + (i%8), n = L%16
//   A 16x32 bf16:  lane L, elem i:  m = L%16,
//                                   k = (i/8)*16 + (L/16)*8 + (i%8)
//   C/D 16x16 f32: lane L, vgpr r:  m = r + 8*(L/16),           n = L%16

__global__ void pack_weights(const float* __restrict__ Wx,
                             const float* __restrict__ Wh,
                             const float* __restrict__ Wout,
                             __bf16* __restrict__ ws) {
    const unsigned total  = WXP_ELEMS + WHP_ELEMS + WOP_ELEMS;
    const unsigned stride = gridDim.x * blockDim.x;
    for (unsigned idx = blockIdx.x * blockDim.x + threadIdx.x; idx < total; idx += stride) {
        float v;
        unsigned dst;
        if (idx < WXP_ELEMS) {                       // Wx: [kt(2)][nt(64)]
            unsigned frag = idx >> 9, e = idx & 511u;
            unsigned kt = frag >> 6, nt = frag & 63u;
            unsigned lane = e >> 4, i = e & 15u;
            unsigned k = (i >> 3) * 16u + (lane >> 4) * 8u + (i & 7u);
            unsigned n = lane & 15u;
            v = Wx[(kt * 32u + k) * G_ + nt * 16u + n];
            dst = WXP_OFF + idx;
        } else if (idx < WXP_ELEMS + WHP_ELEMS) {    // Wh: [g(8)][kt(8)][qj(8)]
            unsigned j2 = idx - WXP_ELEMS;
            unsigned frag = j2 >> 9, e = j2 & 511u;
            unsigned g  = frag >> 6;
            unsigned kt = (frag >> 3) & 7u;
            unsigned qj = frag & 7u;
            unsigned nt = (qj >> 1) * 16u + 2u * g + (qj & 1u);
            unsigned lane = e >> 4, i = e & 15u;
            unsigned k = (i >> 3) * 16u + (lane >> 4) * 8u + (i & 7u);
            unsigned n = lane & 15u;
            v = Wh[(kt * 32u + k) * G_ + nt * 16u + n];
            dst = WHP_OFF + j2;
        } else {                                     // Wout: [kt(8)], cols >= C zero
            unsigned j2 = idx - WXP_ELEMS - WHP_ELEMS;
            unsigned kt = j2 >> 9, e = j2 & 511u;
            unsigned lane = e >> 4, i = e & 15u;
            unsigned k = (i >> 3) * 16u + (lane >> 4) * 8u + (i & 7u);
            unsigned n = lane & 15u;
            v = (n < C_) ? Wout[(kt * 32u + k) * C_ + n] : 0.0f;
            dst = WOP_OFF + j2;
        }
        ws[dst] = (__bf16)v;
    }
}

__global__ void pack_x(const float* __restrict__ x, __bf16* __restrict__ ws) {
    const unsigned stride = gridDim.x * blockDim.x;
    for (unsigned idx = blockIdx.x * blockDim.x + threadIdx.x; idx < XP_ELEMS; idx += stride) {
        unsigned e    = idx & 511u;
        unsigned frag = idx >> 9;
        unsigned kt   = frag & 1u;
        unsigned btt  = frag >> 1;           // bt*1024 + t
        unsigned t    = btt & 1023u;
        unsigned bt   = btt >> 10;
        unsigned lane = e >> 4, i = e & 15u;
        unsigned m = lane & 15u;
        unsigned k = (i >> 3) * 16u + (lane >> 4) * 8u + (i & 7u);
        float v = x[((bt * 16u + m) * (unsigned)T_ + t) * (unsigned)D_ + kt * 32u + k];
        ws[XP_OFF + idx] = (__bf16)v;
    }
}

// h0 = 0 (both parities) and barrier counters = 0: rerun every call so the
// persistent kernel is deterministic across graph replays.
__global__ void init_state(__bf16* __restrict__ ws) {
    unsigned idx = blockIdx.x * blockDim.x + threadIdx.x;
    if (idx < HX_ELEMS) ws[HX_OFF + idx] = (__bf16)0.0f;
    if (idx < 8u) ((unsigned*)((char*)ws + CNT_BYTE))[idx] = 0u;
}

__device__ __forceinline__ float sigf(float x) {
    return 1.0f / (1.0f + __expf(-x));
}
__device__ __forceinline__ float tanh_fast(float x) {
    float e = __expf(2.0f * x);
    return 1.0f - 2.0f / (e + 1.0f);
}

// Opaque wave-uniform global base: SGPR-laundered (defeats LICM; keeps
// global_load saddr form on the LOADcnt-only path).
__device__ __forceinline__ const gbf* glaunder(const __bf16* p) {
    unsigned long long u = (unsigned long long)p;
    asm volatile("" : "+s"(u));
    return (const gbf*)u;
}
__device__ __forceinline__ v16bf gload(const gbf* base, int elem_off) {
    return *(const gv16bf*)(base + elem_off);
}

__launch_bounds__(64, 1)
__global__ void lstm_scan(__bf16* __restrict__ ws,
                          const float* __restrict__ b,
                          const float* __restrict__ bout,
                          float* __restrict__ out) {
    // This WG's Wh slice as ready-to-issue B fragments: [kt(8)][q*2+j(8)] * 512
    __shared__ __align__(32) __bf16 whL[8 * 8 * 512];   // 64 KB

    const int tid  = threadIdx.x;
    const int lane = tid & 31;
    const int j    = tid >> 5;            // wave 0/1: h sub-columns [16j,16j+16) of strip
    const int bt   = blockIdx.x >> 3;     // batch tile
    const int g    = blockIdx.x & 7;      // column group: h strip [32g, 32g+32)
    const int loff = lane * 16;

    // ---- one-time: Wh slice -> LDS (64 KB, contiguous in packed buffer) ----
    {
        const __bf16* src = ws + WHP_OFF + (unsigned)g * (8u * 8u * 512u);
        for (int i = tid; i < 2048; i += 64)
            *(v16bf*)&whL[i * 16] = *(const v16bf*)(src + i * 16);
    }

    // biases: gate q -> column (q*16 + 2g + j)*16 + lane%16
    float bv[4];
    #pragma unroll
    for (int q = 0; q < 4; ++q)
        bv[q] = b[(q * 16 + 2 * g + j) * 16 + (lane & 15)];
    const float boutv = ((lane & 15) < C_) ? bout[lane & 15] : 0.0f;

    v8f cc;
    #pragma unroll
    for (int r = 0; r < 8; ++r) cc[r] = 0.0f;

    unsigned* cnt = (unsigned*)((char*)ws + CNT_BYTE) + bt;
    const __bf16* xP = ws + XP_OFF;

    __syncthreads();   // whL ready

    int p = 0;
    #pragma unroll 1
    for (int t = 0; t < T_; ++t) {
        // opaque per-iteration bases/offsets: keep loads inside the loop
        const gbf* wxp = glaunder(ws + WXP_OFF);
        int zero = 0;
        asm volatile("" : "+v"(zero));

        // ---- gates = bias + x_t @ Wx + h_{t-1} @ Wh ----
        v8f acc[4];
        #pragma unroll
        for (int q = 0; q < 4; ++q)
            #pragma unroll
            for (int r = 0; r < 8; ++r) acc[q][r] = bv[q];

        const __bf16* xfr = xP + ((unsigned)(bt * T_ + t) * 2u) * 512u + (unsigned)loff;
        __builtin_prefetch(xfr + 2 * 512, 0, 3);

        #pragma unroll
        for (int kt = 0; kt < 2; ++kt) {
            v16bf xa = *(const v16bf*)(xfr + kt * 512);
            #pragma unroll
            for (int q = 0; q < 4; ++q) {
                v16bf wb = gload(wxp, (kt * 64 + q * 16 + 2 * g + j) * 512 + loff);
                acc[q] = __builtin_amdgcn_wmma_f32_16x16x32_bf16(
                    false, xa, false, wb, (short)0, acc[q], false, false);
            }
        }

        const __bf16* hx = ws + HX_OFF + (unsigned)((bt * 2 + p) * 8) * 512u;
        #pragma unroll
        for (int kt = 0; kt < 8; ++kt) {
            v16bf ha = *(const v16bf*)(hx + kt * 512 + loff);
            #pragma unroll
            for (int q = 0; q < 4; ++q) {
                v16bf wb = *(const v16bf*)(&whL[zero + (kt * 8 + q * 2 + j) * 512 + loff]);
                acc[q] = __builtin_amdgcn_wmma_f32_16x16x32_bf16(
                    false, ha, false, wb, (short)0, acc[q], false, false);
            }
        }

        // ---- LSTM cell; scatter new h into A-fragment kt=g of HX[bt][1-p] ----
        const int np = p ^ 1;
        __bf16* hdst = ws + HX_OFF + (unsigned)(((bt * 2 + np) * 8 + g)) * 512u;
        #pragma unroll
        for (int r = 0; r < 8; ++r) {
            float iv = sigf(acc[0][r]);
            float fv = sigf(acc[1][r]);
            float gv = tanh_fast(acc[2][r]);
            float ov = sigf(acc[3][r]);
            float cn = fv * cc[r] + iv * gv;
            cc[r] = cn;
            float hv = ov * tanh_fast(cn);
            int m     = r + 8 * (lane >> 4);
            int lanep = m + 16 * ((lane & 15) >> 3);   // klocal = 16j + lane%16
            int ii    = j * 8 + (lane & 7);
            hdst[lanep * 16 + ii] = (__bf16)hv;
        }

        // ---- device-scope barrier across the 8 WGs of this batch group ----
        __threadfence();                       // h stores visible at agent scope
        __syncthreads();
        if (tid == 0) {
            __hip_atomic_fetch_add(cnt, 1u, __ATOMIC_ACQ_REL, __HIP_MEMORY_SCOPE_AGENT);
            const unsigned target = 8u * (unsigned)(t + 1);
            while (__hip_atomic_load(cnt, __ATOMIC_ACQUIRE, __HIP_MEMORY_SCOPE_AGENT) < target)
                __builtin_amdgcn_s_sleep(1);
        }
        __syncthreads();
        __threadfence();                       // invalidate before reading peers' h

        // ---- fused projection: logits_t = h_t @ Wout + bout (WG g=0, wave 0) ----
        if (g == 0 && j == 0) {
            const gbf* wop = glaunder(ws + WOP_OFF);
            const __bf16* hx2 = ws + HX_OFF + (unsigned)((bt * 2 + np) * 8) * 512u;
            v8f pa;
            #pragma unroll
            for (int r = 0; r < 8; ++r) pa[r] = boutv;
            #pragma unroll
            for (int kt = 0; kt < 8; ++kt) {
                v16bf ha = *(const v16bf*)(hx2 + kt * 512 + loff);
                v16bf wb = gload(wop, kt * 512 + loff);
                pa = __builtin_amdgcn_wmma_f32_16x16x32_bf16(
                    false, ha, false, wb, (short)0, pa, false, false);
            }
            int n = lane & 15;
            if (n < C_) {
                #pragma unroll
                for (int r = 0; r < 8; ++r) {
                    int m = r + 8 * (lane >> 4);
                    out[((bt * 16 + m) * T_ + t) * C_ + n] = pa[r];
                }
            }
        }
        p = np;
    }
}

extern "C" void kernel_launch(void* const* d_in, const int* in_sizes, int n_in,
                              void* d_out, int out_size, void* d_ws, size_t ws_size,
                              hipStream_t stream) {
    const float* x    = (const float*)d_in[0];
    const float* Wx   = (const float*)d_in[1];
    const float* Wh   = (const float*)d_in[2];
    const float* b    = (const float*)d_in[3];
    const float* Wout = (const float*)d_in[4];
    const float* bout = (const float*)d_in[5];
    float*  out = (float*)d_out;
    __bf16* ws  = (__bf16*)d_ws;

    const unsigned wtotal = WXP_ELEMS + WHP_ELEMS + WOP_ELEMS;
    pack_weights<<<(wtotal + 255) / 256, 256, 0, stream>>>(Wx, Wh, Wout, ws);
    pack_x<<<8192, 256, 0, stream>>>(x, ws);
    init_state<<<(HX_ELEMS + 255) / 256, 256, 0, stream>>>(ws);
    lstm_scan<<<64, 64, 0, stream>>>(ws, b, bout, out);
}
